// HyperGCNconv_84980222918798
// MI455X (gfx1250) — matrix operand order repaired and checked by reading, not compiled, not run on hardware
//
#include <hip/hip_runtime.h>
#include <hip/hip_bf16.h>

typedef __attribute__((ext_vector_type(16))) _Float16 v16h;
typedef __attribute__((ext_vector_type(8)))  float    v8f;

constexpr int       kNodes = 100000;
constexpr int       kEdges = 20000;
constexpr long long kNNZ   = 2000000;
constexpr int       kDIn   = 128;
constexpr int       kDOut  = 128;

// ---------------------------------------------------------------------------
// zero-fill (grid-stride)
// ---------------------------------------------------------------------------
__global__ void zero_kernel(float* __restrict__ p, size_t n) {
    size_t i      = (size_t)blockIdx.x * blockDim.x + threadIdx.x;
    size_t stride = (size_t)gridDim.x * blockDim.x;
    for (; i < n; i += stride) p[i] = 0.0f;
}

// ---------------------------------------------------------------------------
// xl = x @ W^T + b via v_wmma_f32_16x16x32_f16
// one wave -> one 16x16 tile; 8 waves/block -> all 128 out channels
// ---------------------------------------------------------------------------
__device__ __forceinline__ void cvt8(const float* __restrict__ p, _Float16* d) {
    float4 u = *reinterpret_cast<const float4*>(p);
    float4 v = *reinterpret_cast<const float4*>(p + 4);
    d[0] = (_Float16)u.x; d[1] = (_Float16)u.y;
    d[2] = (_Float16)u.z; d[3] = (_Float16)u.w;
    d[4] = (_Float16)v.x; d[5] = (_Float16)v.y;
    d[6] = (_Float16)v.z; d[7] = (_Float16)v.w;
}

__global__ __launch_bounds__(256)
void gemm_wmma_kernel(const float* __restrict__ x,
                      const float* __restrict__ W,
                      const float* __restrict__ bias,
                      float* __restrict__ xl) {
    const int lane = threadIdx.x & 31;
    const int wave = threadIdx.x >> 5;        // 0..7 -> out-channel tile
    const int row0 = blockIdx.x * 16;         // 16 node rows per block
    const int col0 = wave * 16;               // 16 out channels per wave
    const int hs   = lane >> 4;               // K half-select (0/1)
    const int m    = row0 + (lane & 15);      // A row this lane feeds
    const int o    = col0 + (lane & 15);      // B column (= W row) this lane feeds

    v8f c = {};
    union { v16h v; _Float16 e[16]; } a, b;

#pragma unroll
    for (int kc = 0; kc < kDIn; kc += 32) {
        // A (16x32 f16): lane holds K = kc+8*hs..+7 and kc+16+8*hs..+7 of row m
        cvt8(x + (size_t)m * kDIn + kc + 8 * hs,      a.e);
        cvt8(x + (size_t)m * kDIn + kc + 16 + 8 * hs, a.e + 8);
        // B (32x16 f16): lane holds K = kc+8*hs..+7 and kc+16+8*hs..+7 of column o; B[k][o] = W[o][k]
        cvt8(W + (size_t)o * kDIn + kc + 8 * hs,      b.e);
        cvt8(W + (size_t)o * kDIn + kc + 16 + 8 * hs, b.e + 8);
        // (neg_a, A, neg_b, B, c_mod, C, reuse_a, reuse_b)
        c = __builtin_amdgcn_wmma_f32_16x16x32_f16(false, a.v, false, b.v,
                                                   (short)0, c, false, false);
    }

    const float bv = bias[o];
#pragma unroll
    for (int j = 0; j < 8; ++j) {
        const int row = row0 + j + 8 * hs;    // C/D layout: VGPR j -> M = j + 8*(lane/16)
        xl[(size_t)row * kDOut + o] = c[j] + bv;
    }
}

// ---------------------------------------------------------------------------
// D[v] += HEW[e], Bc[e] += 1  (per-nnz scalar atomics)
// ---------------------------------------------------------------------------
__global__ void degree_kernel(const long long* __restrict__ HE,
                              const float* __restrict__ HEW,
                              float* __restrict__ Dv,
                              float* __restrict__ Bc) {
    long long i = (long long)blockIdx.x * blockDim.x + threadIdx.x;
    if (i >= kNNZ) return;
    const int v = (int)HE[i];
    const int e = (int)HE[kNNZ + i];
    atomicAdd(&Dv[v], HEW[e]);
    atomicAdd(&Bc[e], 1.0f);
}

// in-place D -> Dinv, Bc -> Binv
__global__ void inv_kernel(float* __restrict__ Dv, float* __restrict__ Bc) {
    const int i = blockIdx.x * blockDim.x + threadIdx.x;
    if (i < kNodes) { const float d = Dv[i]; Dv[i] = d > 0.f ? 1.f / d : 0.f; }
    if (i < kEdges) { const float c = Bc[i]; Bc[i] = c > 0.f ? 1.f / c : 0.f; }
}

// ---------------------------------------------------------------------------
// e_acc[e,:] += xl[v,:]   (one 32-lane wave per nnz, float4 per lane)
// ---------------------------------------------------------------------------
__global__ __launch_bounds__(256)
void scatter_edge_kernel(const long long* __restrict__ HE,
                         const float* __restrict__ xl,
                         float* __restrict__ e_acc) {
    const long long idx = (long long)blockIdx.x * 8 + (threadIdx.x >> 5);
    if (idx >= kNNZ) return;
    const int lane = threadIdx.x & 31;
    const int v = (int)HE[idx];
    const int e = (int)HE[kNNZ + idx];
    const float4 val = *reinterpret_cast<const float4*>(xl + (size_t)v * kDOut + lane * 4);
    float* dst = e_acc + (size_t)e * kDOut + lane * 4;
    atomicAdd(dst + 0, val.x);
    atomicAdd(dst + 1, val.y);
    atomicAdd(dst + 2, val.z);
    atomicAdd(dst + 3, val.w);
}

// ---------------------------------------------------------------------------
// out[v,:] += Dinv[v]*Binv[e] * e_acc[e,:]
// ---------------------------------------------------------------------------
__global__ __launch_bounds__(256)
void scatter_node_kernel(const long long* __restrict__ HE,
                         const float* __restrict__ e_acc,
                         const float* __restrict__ Dinv,
                         const float* __restrict__ Binv,
                         float* __restrict__ out) {
    const long long idx = (long long)blockIdx.x * 8 + (threadIdx.x >> 5);
    if (idx >= kNNZ) return;
    const int lane = threadIdx.x & 31;
    const int v = (int)HE[idx];
    const int e = (int)HE[kNNZ + idx];
    const float s = Dinv[v] * Binv[e];
    const float4 val = *reinterpret_cast<const float4*>(e_acc + (size_t)e * kDOut + lane * 4);
    float* dst = out + (size_t)v * kDOut + lane * 4;
    atomicAdd(dst + 0, s * val.x);
    atomicAdd(dst + 1, s * val.y);
    atomicAdd(dst + 2, s * val.z);
    atomicAdd(dst + 3, s * val.w);
}

// ---------------------------------------------------------------------------
extern "C" void kernel_launch(void* const* d_in, const int* in_sizes, int n_in,
                              void* d_out, int out_size, void* d_ws, size_t ws_size,
                              hipStream_t stream) {
    const float*     x    = (const float*)d_in[0];
    const long long* HE   = (const long long*)d_in[1];   // int64 [2, nnz]
    const float*     HEW  = (const float*)d_in[2];
    const float*     W    = (const float*)d_in[3];
    const float*     bias = (const float*)d_in[4];
    float*           out  = (float*)d_out;
    float*           ws   = (float*)d_ws;

    // workspace layout (floats): D | Bc | e_acc | xl   (~62 MB total)
    float* Dv    = ws;
    float* Bc    = Dv + kNodes;
    float* e_acc = Bc + kEdges;
    float* xl    = e_acc + (size_t)kEdges * kDOut;

    const size_t zero_front = (size_t)kNodes + kEdges + (size_t)kEdges * kDOut;
    zero_kernel<<<2048, 256, 0, stream>>>(ws, zero_front);
    zero_kernel<<<4096, 256, 0, stream>>>(out, (size_t)kNodes * kDOut);

    gemm_wmma_kernel<<<kNodes / 16, 256, 0, stream>>>(x, W, bias, xl);

    degree_kernel<<<(unsigned)((kNNZ + 255) / 256), 256, 0, stream>>>(HE, HEW, Dv, Bc);
    inv_kernel<<<(kNodes + 255) / 256, 256, 0, stream>>>(Dv, Bc);

    scatter_edge_kernel<<<(unsigned)((kNNZ + 7) / 8), 256, 0, stream>>>(HE, xl, e_acc);
    scatter_node_kernel<<<(unsigned)((kNNZ + 7) / 8), 256, 0, stream>>>(HE, e_acc, Dv, Bc, out);
}
